// TemporalCorrelation_49770081026663
// MI455X (gfx1250) — matrix operand order repaired and checked by reading, not compiled, not run on hardware
//
#include <hip/hip_runtime.h>
#include <hip/hip_bf16.h>

// ---------------------------------------------------------------------------
// TemporalCorrelation for MI455X (gfx1250, wave32, WMMA) — single kernel.
//   feat1, feat2 : [B, C, T] fp32   (B=8, C=1024, T=4096)
//   out          : [B, 21, T] fp32  (21 = 2*MAX_DISP+1, MAX_DISP=10)
//
// corr[b,o,t] = <f1[:,t], f2[:,t']> * inv1[t] * inv2[t'],  t' = clamp(t+o-10)
// i.e. normalization is a per-column post-scale of the raw dot products, so
// the whole problem is the width-21 band of F1^T * F2 (per batch), computed
// with v_wmma_f32_16x16x32_f16, with column sum-of-squares accumulated from
// the same streamed values (no separate normalization pass -> ~1x HBM read).
// ---------------------------------------------------------------------------

#define TC_B 8
#define TC_C 1024
#define TC_T 4096
#define TC_MAXD 10
#define TC_NOFF (2 * TC_MAXD + 1)   // 21

typedef __attribute__((ext_vector_type(16))) _Float16 v16h;
typedef __attribute__((ext_vector_type(8)))  float    v8f;

union V16H {
    v16h v;
    _Float16 e[16];
};

__device__ __forceinline__ int clampi(int x, int lo, int hi) {
    return x < lo ? lo : (x > hi ? hi : x);
}

__device__ __forceinline__ float inv_norm(float s) {
    // matches torch F.normalize: x / max(||x||, 1e-12)
    return 1.0f / fmaxf(sqrtf(s), 1e-12f);
}

// ---------------------------------------------------------------------------
// One wave32 per (b, 16-wide t block).  D(16x48) = A(16xC) * [B0|B1|B2](Cx48),
// K swept 32 at a time; band: out[b,o,t0+m] = D[m][m+o+6] * invA[m] * invB[n].
// Fragment element->lane mapping follows CDNA5 ISA §7.12.2:
//   A 16x32 : lane L: M=L%16, K = 8*(L/16)+{0..7} and 16+8*(L/16)+{0..7}
//   B 32x16 : lane L: N=L%16, K = 8*(L/16)+{0..7} and 16+8*(L/16)+{0..7}
// Lanes L and L+16 jointly cover all K, so column sums-of-squares are the
// per-lane running sums combined with one shfl_xor(16).
// ---------------------------------------------------------------------------
__global__ __launch_bounds__(32) void tc_corr_wmma_kernel(
        const float* __restrict__ f1, const float* __restrict__ f2,
        float* __restrict__ out) {
    __shared__ float dtile[3 * 16 * 16];   // three 16x16 f32 accumulator tiles
    __shared__ float invA_lds[16];         // inv norms for f1 columns t0+0..15
    __shared__ float invB_lds[48];         // inv norms for f2 columns t0-16..t0+31

    const int lane = threadIdx.x;          // 0..31 (wave32)
    const int lo   = lane & 15;
    const int hi   = lane >> 4;            // 0 or 1
    const int t0   = blockIdx.x * 16;
    const int b    = blockIdx.y;

    const size_t baseIn = (size_t)b * TC_C * TC_T;

    // ---- A-matrix lane geometry: M = lo (t index).
    const int    tA = t0 + lo;                         // always in range
    const float* pA = f1 + baseIn + tA;
    const int    kA = hi * 8;                          // K base for this lane half

    // ---- B-matrix lane geometry per N-tile j: N = lo, t2 = clamp(t0-16+16j+lo).
    const int tt0 = clampi(t0 - 16 + lo, 0, TC_T - 1);
    const int tt1 = clampi(t0      + lo, 0, TC_T - 1);
    const int tt2 = clampi(t0 + 16 + lo, 0, TC_T - 1);
    const float* pB0 = f2 + baseIn + tt0;
    const float* pB1 = f2 + baseIn + tt1;
    const float* pB2 = f2 + baseIn + tt2;
    const int kB = hi * 8;                             // K base for this lane half

    v8f acc0 = {}, acc1 = {}, acc2 = {};
    float sqA = 0.0f, sq0 = 0.0f, sq1 = 0.0f, sq2 = 0.0f;

    for (int k0 = 0; k0 < TC_C; k0 += 32) {
        // Build A fragment and accumulate raw sum-of-squares for column tA.
        V16H a;
        #pragma unroll
        for (int i = 0; i < 8; ++i) {
            float x0 = pA[(size_t)(k0 + kA + i)      * TC_T];
            float x1 = pA[(size_t)(k0 + 16 + kA + i) * TC_T];
            sqA = fmaf(x0, x0, sqA);
            sqA = fmaf(x1, x1, sqA);
            a.e[i]     = (_Float16)x0;
            a.e[i + 8] = (_Float16)x1;
        }
        // Build B fragments (K = 8*h+{0..7} and 16+8*h+{0..7}, same as A) + sums-of-squares.
        V16H b0, b1, b2;
        #pragma unroll
        for (int i = 0; i < 16; ++i) {
            size_t off = (size_t)(k0 + kB + i + (i & 8)) * TC_T;
            float x0 = pB0[off];
            float x1 = pB1[off];
            float x2 = pB2[off];
            sq0 = fmaf(x0, x0, sq0);
            sq1 = fmaf(x1, x1, sq1);
            sq2 = fmaf(x2, x2, sq2);
            b0.e[i] = (_Float16)x0;
            b1.e[i] = (_Float16)x1;
            b2.e[i] = (_Float16)x2;
        }
        // D += A x Bj   (args: neg_a, A, neg_b, B, c_mod, C, reuse_a, reuse_b)
        acc0 = __builtin_amdgcn_wmma_f32_16x16x32_f16(false, a.v, false, b0.v,
                                                      (short)0, acc0, false, false);
        acc1 = __builtin_amdgcn_wmma_f32_16x16x32_f16(false, a.v, false, b1.v,
                                                      (short)0, acc1, false, false);
        acc2 = __builtin_amdgcn_wmma_f32_16x16x32_f16(false, a.v, false, b2.v,
                                                      (short)0, acc2, false, false);
    }

    // Combine lane-half partial sums: lanes L and L+16 cover complementary K.
    sqA += __shfl_xor(sqA, 16, 32);
    sq0 += __shfl_xor(sq0, 16, 32);
    sq1 += __shfl_xor(sq1, 16, 32);
    sq2 += __shfl_xor(sq2, 16, 32);

    if (lane < 16) {
        invA_lds[lo]      = inv_norm(sqA);
        invB_lds[lo]      = inv_norm(sq0);
        invB_lds[16 + lo] = inv_norm(sq1);
        invB_lds[32 + lo] = inv_norm(sq2);
    }

    // Dump accumulators to LDS. C/D layout: lane L holds N=L%16, M=8*(L/16)+r.
    #pragma unroll
    for (int r = 0; r < 8; ++r) {
        int m = hi * 8 + r;
        dtile[0 * 256 + m * 16 + lo] = acc0[r];
        dtile[1 * 256 + m * 16 + lo] = acc1[r];
        dtile[2 * 256 + m * 16 + lo] = acc2[r];
    }
    __syncthreads();   // single-wave WG: orders LDS (barrier lowers to s_nop)

    // Band extraction + normalization: out[b,o,t0+m] = D[m][m+o+6]*invA*invB
    for (int idx = lane; idx < TC_NOFF * 16; idx += 32) {
        int o   = idx >> 4;
        int m   = idx & 15;
        int n48 = m + o + 6;                 // in [6, 41]
        float v = dtile[(n48 >> 4) * 256 + m * 16 + (n48 & 15)];
        v *= invA_lds[m] * invB_lds[n48];
        out[((size_t)b * TC_NOFF + o) * TC_T + t0 + m] = v;
    }
}

// ---------------------------------------------------------------------------
extern "C" void kernel_launch(void* const* d_in, const int* in_sizes, int n_in,
                              void* d_out, int out_size, void* d_ws, size_t ws_size,
                              hipStream_t stream) {
    (void)in_sizes; (void)n_in; (void)out_size; (void)d_ws; (void)ws_size;

    const float* feat1 = (const float*)d_in[0];
    const float* feat2 = (const float*)d_in[1];
    float*       out   = (float*)d_out;

    // One wave32 per (b, 16-wide t block): grid 256 x 8, fused norm+correlate.
    dim3 grid(TC_T / 16, TC_B);
    tc_corr_wmma_kernel<<<grid, 32, 0, stream>>>(feat1, feat2, out);
}
